// SSM_black_con_42356967473609
// MI455X (gfx1250) — matrix operand order repaired and checked by reading, not compiled, not run on hardware
//
#include <hip/hip_runtime.h>
#include <hip/hip_bf16.h>
#include <stdint.h>

#define T_DIM 512
#define B_DIM 256
#define NX    256
#define NU    256
#define ND_   128
#define NM_   128
#define NDT_  128
#define NH    512
#define S_CH  8          // scan chunk stride: 8 independent chains, 64 steps each

typedef __attribute__((ext_vector_type(16))) __bf16 v16bf;
typedef __attribute__((ext_vector_type(8)))  float  v8f;

// ---------------- WMMA fragment helpers (wave32, 16x16x32 bf16) ----------------
// A-matrix 16x32 (MxK), 16-bit: lane L holds row m=L&15; kbase=(L>=16)?8:0;
// VGPR v: k = kbase + (v>=4?16:0) + 2*(v&3) (packed pair k,k+1).
__device__ __forceinline__ v16bf load_a_frag_lds(const __bf16* src, int ldm, int k0, int lane) {
  int m  = lane & 15;
  int kb = (lane >> 4) * 8;
  v16bf a;
#pragma unroll
  for (int v = 0; v < 8; ++v) {
    int k = k0 + kb + ((v >> 2) * 16) + ((v & 3) * 2);
    a[2 * v]     = src[m * ldm + k];
    a[2 * v + 1] = src[m * ldm + k + 1];
  }
  return a;
}

// B-matrix 32x16 (KxN), 16-bit: lane L holds col n=L&15; kbase=(L>=16)?8:0;
// VGPR v: k = kbase + (v>=4?16:0) + 2*(v&3). Source W row-major (N,K): B[k][n] = W[n0+n][k]; W pre-offset by k0.
__device__ __forceinline__ v16bf load_b_frag_w(const float* __restrict__ W, int ldk, int n0, int lane) {
  int n  = lane & 15;
  int kb = (lane >> 4) * 8;
  const float* p = W + (n0 + n) * ldk + kb;
  v16bf b;
#pragma unroll
  for (int v = 0; v < 8; ++v) {
    int k = ((v >> 2) * 16) + ((v & 3) * 2);
    b[2 * v]     = (__bf16)p[k];
    b[2 * v + 1] = (__bf16)p[k + 1];
  }
  return b;
}

__device__ __forceinline__ v8f wmma_bf16(v16bf a, v16bf b, v8f c) {
  return __builtin_amdgcn_wmma_f32_16x16x32_bf16(false, a, false, b, (short)0, c, false, false);
}

// ---------------- K1: G1[n][k] = A_effT = (1 - 0.1*sigmoid(As)) * softmax_row(Aw)
__global__ void aeff_kernel(const float* __restrict__ Aw, const float* __restrict__ As,
                            float* __restrict__ G1) {
  __shared__ float red[256];
  int n = blockIdx.x, k = threadIdx.x;
  float w = Aw[n * NX + k];
  red[k] = w; __syncthreads();
#pragma unroll
  for (int s = 128; s > 0; s >>= 1) { if (k < s) red[k] = fmaxf(red[k], red[k + s]); __syncthreads(); }
  float mx = red[0]; __syncthreads();
  float e = __expf(w - mx);
  red[k] = e; __syncthreads();
#pragma unroll
  for (int s = 128; s > 0; s >>= 1) { if (k < s) red[k] += red[k + s]; __syncthreads(); }
  float sm = e / red[0];
  float sg = 1.0f / (1.0f + __expf(-As[n * NX + k]));
  G1[n * NX + k] = (1.0f - 0.1f * sg) * sm;
}

// ---------------- K2: A-power chain in f32: G_{p+1} = G_p @ G_1 (powers commute)
__global__ void matmul256_kernel(const float* __restrict__ Ga, const float* __restrict__ Gb,
                                 float* __restrict__ Gc) {
  __shared__ float arow[256];
  int n = blockIdx.x, k = threadIdx.x;
  arow[k] = Ga[n * 256 + k];
  __syncthreads();
  float acc = 0.0f;
  for (int m = 0; m < 256; ++m) acc = fmaf(arow[m], Gb[m * 256 + k], acc);
  Gc[n * 256 + k] = acc;
}

// ---------------- K3: pack a row-major (N,K) f32 weight into fragment-ready bf16
// layout: out[(ntile*nkblk + kb)*32 + lane] = v16bf fragment (pure b128 loads later)
__global__ void pack_b_kernel(const float* __restrict__ W, int ldk, int nkblk,
                              v16bf* __restrict__ out) {
  int frag  = blockIdx.x;            // ntile*nkblk + kb
  int kb    = frag % nkblk;
  int ntile = frag / nkblk;
  out[frag * 32 + threadIdx.x] = load_b_frag_w(W + kb * 32, ldk, ntile * 16, threadIdx.x);
}

// ---------------- K4: fused MLP (16 waves / 16-row block of T*B)
__global__ void __launch_bounds__(512)
mlp_kernel(const float* __restrict__ Mfl, const float* __restrict__ DTin,
           const float* __restrict__ Din, const float* __restrict__ UMIN,
           const float* __restrict__ UMAX, const v16bf* __restrict__ hf1p,
           const v16bf* __restrict__ hf2p, const v16bf* __restrict__ BWp,
           const v16bf* __restrict__ EWp, float* __restrict__ Sumin,
           float* __restrict__ Sumax, float* __restrict__ drive) {
  __shared__ __bf16 MUs[16 * 256];   // 8 KB
  __shared__ __bf16 Ds [16 * 128];   // 4 KB
  __shared__ __bf16 Hs [16 * NH];    // 16 KB
  __shared__ __bf16 Us [16 * NU];    // 8 KB
  const int lane = threadIdx.x & 31;
  const int wave = threadIdx.x >> 5;
  const int hi   = lane >> 4;
  const int n    = lane & 15;
  const int r0   = blockIdx.x * 16;

  // stage A-side once per block (convert f32->bf16 once, not per wave)
  for (int i = threadIdx.x; i < 16 * 256; i += 512) {
    int m = i >> 8, k = i & 255;
    float f = (k < NM_) ? Mfl[(r0 + m) * NM_ + k] : DTin[(r0 + m) * NDT_ + (k - NM_)];
    MUs[i] = (__bf16)f;
  }
  for (int i = threadIdx.x; i < 16 * 128; i += 512) {
    int m = i >> 7, k = i & 127;
    Ds[i] = (__bf16)Din[(r0 + m) * ND_ + k];
  }
  __syncthreads();

  // Phase 1: H = relu(MU @ hf1^T)  (each wave: two 16-col tiles of NH)
#pragma unroll
  for (int half = 0; half < 2; ++half) {
    int nt = wave * 2 + half;
    v8f c = {};
#pragma unroll
    for (int kb = 0; kb < 8; ++kb) {
      v16bf a = load_a_frag_lds(MUs, 256, kb * 32, lane);
      c = wmma_bf16(a, hf1p[(nt * 8 + kb) * 32 + lane], c);
    }
#pragma unroll
    for (int v = 0; v < 8; ++v)
      Hs[(v + hi * 8) * NH + nt * 16 + n] = (__bf16)fmaxf(c[v], 0.0f);
  }
  __syncthreads();

  // Phase 2: u = relu(H @ hf2^T), K = 512
  v8f cu = {};
#pragma unroll
  for (int kb = 0; kb < 16; ++kb) {
    v16bf a = load_a_frag_lds(Hs, NH, kb * 32, lane);
    cu = wmma_bf16(a, hf2p[(wave * 16 + kb) * 32 + lane], cu);
  }
#pragma unroll
  for (int v = 0; v < 8; ++v) {
    int m  = v + hi * 8;
    int gi = (r0 + m) * NU + wave * 16 + n;
    float uv = fmaxf(cu[v], 0.0f);
    Sumin[gi] = fmaxf(UMIN[gi] - uv, 0.0f);
    Sumax[gi] = fmaxf(uv - UMAX[gi], 0.0f);
    Us[m * NU + wave * 16 + n] = (__bf16)uv;
  }
  __syncthreads();

  // Phase 3: drive = U@B_W^T (K=256) + D@E_W^T (K=128)
  v8f cd = {};
#pragma unroll
  for (int kb = 0; kb < 8; ++kb) {
    v16bf a = load_a_frag_lds(Us, NU, kb * 32, lane);
    cd = wmma_bf16(a, BWp[(wave * 8 + kb) * 32 + lane], cd);
  }
#pragma unroll
  for (int kb = 0; kb < 4; ++kb) {
    v16bf a = load_a_frag_lds(Ds, 128, kb * 32, lane);
    cd = wmma_bf16(a, EWp[(wave * 4 + kb) * 32 + lane], cd);
  }
#pragma unroll
  for (int v = 0; v < 8; ++v)
    drive[(r0 + (v + hi * 8)) * NX + wave * 16 + n] = cd[v];
}

// ---------------- K5: E_t = sum_{j=0..6} d_{t+j} @ A^{7-j}  + d_{t+7}
// fully parallel GEMM (K_eff = 1792) that buys 8x parallelism in the scan
__global__ void __launch_bounds__(512)
egemm_kernel(const float* __restrict__ drive, const v16bf* __restrict__ Gp,
             float* __restrict__ E) {
  __shared__ __bf16 ds[7 * 16 * 256];   // 56 KB: rows t..t+6 of the 16-row b-slab
  const int t    = blockIdx.x >> 4;
  const int b0   = (blockIdx.x & 15) << 4;
  const int lane = threadIdx.x & 31;
  const int wave = threadIdx.x >> 5;
  const int hi   = lane >> 4;
  const int n    = lane & 15;

  for (int i = threadIdx.x; i < 7 * 16 * 256; i += 512) {
    int j = i >> 12, m = (i >> 8) & 15, k = i & 255;
    ds[i] = (__bf16)drive[((t + j) * B_DIM + b0 + m) * NX + k];
  }
  __syncthreads();

  v8f c = {};
#pragma unroll
  for (int j = 0; j < 7; ++j) {
    const v16bf* g = Gp + (size_t)(6 - j) * (16 * 8 * 32);   // power 7-j lives at slot (7-j)-1
#pragma unroll
    for (int kb = 0; kb < 8; ++kb) {
      v16bf a = load_a_frag_lds(ds + j * 4096, 256, kb * 32, lane);
      c = wmma_bf16(a, g[(wave * 8 + kb) * 32 + lane], c);
    }
  }
#pragma unroll
  for (int v = 0; v < 8; ++v) {
    int m = v + hi * 8;
    E[(t * B_DIM + b0 + m) * NX + wave * 16 + n] =
        c[v] + drive[((t + 7) * B_DIM + b0 + m) * NX + wave * 16 + n];
  }
}

// ---------------- K6: ramp — first S_CH sequential steps (s_1..s_8 -> X_0..X_7)
__global__ void __launch_bounds__(512)
ramp_kernel(const float* __restrict__ x0, const v16bf* __restrict__ G1p,
            const float* __restrict__ drive, float* __restrict__ X) {
  __shared__ __bf16 xs[16 * NX];
  const int lane = threadIdx.x & 31;
  const int wave = threadIdx.x >> 5;
  const int hi   = lane >> 4;
  const int n    = lane & 15;
  const int b0   = blockIdx.x * 16;
  const int n0   = wave * 16;

  for (int i = threadIdx.x; i < 16 * NX; i += 512)
    xs[i] = (__bf16)x0[(b0 + (i >> 8)) * NX + (i & 255)];
  v16bf bf[8];
#pragma unroll
  for (int kb = 0; kb < 8; ++kb) bf[kb] = G1p[(wave * 8 + kb) * 32 + lane];
  __syncthreads();

  for (int t = 0; t < S_CH; ++t) {
    v8f c = {};
#pragma unroll
    for (int kb = 0; kb < 8; ++kb)
      c = wmma_bf16(load_a_frag_lds(xs, NX, kb * 32, lane), bf[kb], c);
    float xn[8];
#pragma unroll
    for (int v = 0; v < 8; ++v) {
      int gi = (t * B_DIM + b0 + v + hi * 8) * NX + n0 + n;
      xn[v] = c[v] + drive[gi];
      X[gi] = xn[v];
    }
    __syncthreads();
#pragma unroll
    for (int v = 0; v < 8; ++v) xs[(v + hi * 8) * NX + n0 + n] = (__bf16)xn[v];
    __syncthreads();
  }
}

// ---------------- K7: 8 independent chains per b-slab: s_{t+8} = s_t @ A^8 + E_t
__global__ void __launch_bounds__(512)
chain_kernel(const v16bf* __restrict__ G8p, const float* __restrict__ E,
             float* __restrict__ X) {
  __shared__ __bf16 xs[16 * NX];
  const int lane = threadIdx.x & 31;
  const int wave = threadIdx.x >> 5;
  const int hi   = lane >> 4;
  const int n    = lane & 15;
  const int slab = blockIdx.x >> 3;
  const int r    = (blockIdx.x & 7) + 1;     // chain id 1..8, start state s_r = X_{r-1}
  const int b0   = slab * 16;
  const int n0   = wave * 16;

  for (int i = threadIdx.x; i < 16 * NX; i += 512)
    xs[i] = (__bf16)X[((r - 1) * B_DIM + b0 + (i >> 8)) * NX + (i & 255)];
  v16bf bf[8];
#pragma unroll
  for (int kb = 0; kb < 8; ++kb) bf[kb] = G8p[(wave * 8 + kb) * 32 + lane];
  __syncthreads();

  for (int t = r; t <= T_DIM - S_CH; t += S_CH) {   // 63 iterations per chain
    if (t + S_CH <= T_DIM - S_CH)
      __builtin_prefetch(E + ((t + S_CH) * B_DIM + b0) * NX + n0, 0, 1);
    v8f c = {};
#pragma unroll
    for (int kb = 0; kb < 8; ++kb)
      c = wmma_bf16(load_a_frag_lds(xs, NX, kb * 32, lane), bf[kb], c);
    float xn[8];
#pragma unroll
    for (int v = 0; v < 8; ++v) {
      int m  = v + hi * 8;
      float xv = c[v] + E[(t * B_DIM + b0 + m) * NX + n0 + n];
      X[((t + S_CH - 1) * B_DIM + b0 + m) * NX + n0 + n] = xv;
      xn[v] = xv;
    }
    __syncthreads();
#pragma unroll
    for (int v = 0; v < 8; ++v) xs[(v + hi * 8) * NX + n0 + n] = (__bf16)xn[v];
    __syncthreads();
  }
}

// ---------------- K8: fully-parallel slacks + Y at full HBM bandwidth (float4)
__global__ void slack_kernel(const float* __restrict__ X, const float* __restrict__ XMIN,
                             const float* __restrict__ XMAX, float* __restrict__ Sxmin,
                             float* __restrict__ Sxmax, float* __restrict__ Y) {
  int i = blockIdx.x * blockDim.x + threadIdx.x;    // one float4 per thread
  float4 xv = ((const float4*)X)[i];
  float4 mn = ((const float4*)XMIN)[i];
  float4 mx = ((const float4*)XMAX)[i];
  float4 a, b;
  a.x = fmaxf(mn.x - xv.x, 0.0f); a.y = fmaxf(mn.y - xv.y, 0.0f);
  a.z = fmaxf(mn.z - xv.z, 0.0f); a.w = fmaxf(mn.w - xv.w, 0.0f);
  b.x = fmaxf(xv.x - mx.x, 0.0f); b.y = fmaxf(xv.y - mx.y, 0.0f);
  b.z = fmaxf(xv.z - mx.z, 0.0f); b.w = fmaxf(xv.w - mx.w, 0.0f);
  ((float4*)Sxmin)[i] = a;
  ((float4*)Sxmax)[i] = b;
  if ((i & 63) == 63) Y[i >> 6] = xv.w;             // column NX-1
}

extern "C" void kernel_launch(void* const* d_in, const int* in_sizes, int n_in,
                              void* d_out, int out_size, void* d_ws, size_t ws_size,
                              hipStream_t stream) {
  (void)in_sizes; (void)n_in; (void)out_size; (void)ws_size;
  const float* x      = (const float*)d_in[0];
  const float* M_flow = (const float*)d_in[1];
  const float* DT     = (const float*)d_in[2];
  const float* D      = (const float*)d_in[3];
  const float* XMIN   = (const float*)d_in[4];
  const float* XMAX   = (const float*)d_in[5];
  const float* UMIN   = (const float*)d_in[6];
  const float* UMAX   = (const float*)d_in[7];
  const float* A_w    = (const float*)d_in[8];
  const float* A_s    = (const float*)d_in[9];
  const float* B_W    = (const float*)d_in[10];
  const float* E_W    = (const float*)d_in[11];
  const float* hf1    = (const float*)d_in[12];
  const float* hf2    = (const float*)d_in[13];

  // outputs (X, Y, Sx_min, Sx_max, Su_min, Su_max) flat-concatenated
  float* out   = (float*)d_out;
  float* X     = out;
  float* Y     = X + (size_t)T_DIM * B_DIM * NX;
  float* Sxmin = Y + (size_t)T_DIM * B_DIM;
  float* Sxmax = Sxmin + (size_t)T_DIM * B_DIM * NX;
  float* Sumin = Sxmax + (size_t)T_DIM * B_DIM * NX;
  float* Sumax = Sumin + (size_t)T_DIM * B_DIM * NU;

  // workspace: G powers (2MB) | drive (128MB) | E (128MB) | packed bf16 frags (~1.8MB)
  float* G     = (float*)d_ws;                        // G[p-1] at G + (p-1)*65536
  float* drive = G + (size_t)8 * NX * NX;
  float* E     = drive + (size_t)T_DIM * B_DIM * NX;
  v16bf* packB = (v16bf*)(E + (size_t)T_DIM * B_DIM * NX);   // 32B-aligned
  v16bf* hf1p  = packB;                // 32 tiles * 8 kb
  v16bf* hf2p  = hf1p + 32 * 8 * 32;   // 16 tiles * 16 kb
  v16bf* BWp   = hf2p + 16 * 16 * 32;  // 16 * 8
  v16bf* EWp   = BWp + 16 * 8 * 32;    // 16 * 4
  v16bf* Gp    = EWp + 16 * 4 * 32;    // 8 powers * (16*8) frags

  aeff_kernel<<<NX, 256, 0, stream>>>(A_w, A_s, G);
  for (int p = 1; p < 8; ++p)          // G_{p+1} = G_p @ G_1 (f32)
    matmul256_kernel<<<256, 256, 0, stream>>>(G + (size_t)(p - 1) * NX * NX, G,
                                              G + (size_t)p * NX * NX);
  pack_b_kernel<<<32 * 8, 32, 0, stream>>>(hf1, NM_ + NDT_, 8, hf1p);
  pack_b_kernel<<<16 * 16, 32, 0, stream>>>(hf2, NH, 16, hf2p);
  pack_b_kernel<<<16 * 8, 32, 0, stream>>>(B_W, NU, 8, BWp);
  pack_b_kernel<<<16 * 4, 32, 0, stream>>>(E_W, ND_, 4, EWp);
  for (int p = 0; p < 8; ++p)
    pack_b_kernel<<<16 * 8, 32, 0, stream>>>(G + (size_t)p * NX * NX, NX, 8,
                                             Gp + (size_t)p * 16 * 8 * 32);

  mlp_kernel<<<(T_DIM * B_DIM) / 16, 512, 0, stream>>>(
      M_flow, DT, D, UMIN, UMAX, hf1p, hf2p, BWp, EWp, Sumin, Sumax, drive);
  egemm_kernel<<<(T_DIM - S_CH + 1) * 16, 512, 0, stream>>>(drive, Gp, E);
  ramp_kernel<<<B_DIM / 16, 512, 0, stream>>>(x, Gp, drive, X);
  chain_kernel<<<(B_DIM / 16) * S_CH, 512, 0, stream>>>(Gp + (size_t)7 * 16 * 8 * 32, E, X);
  slack_kernel<<<(T_DIM * B_DIM * NX / 4) / 256, 256, 0, stream>>>(
      X, XMIN, XMAX, Sxmin, Sxmax, Y);
}
